// RewardGuidanceModel_17489106830097
// MI455X (gfx1250) — matrix operand order repaired and checked by reading, not compiled, not run on hardware
//
#include <hip/hip_runtime.h>
#include <hip/hip_bf16.h>

#define B_TOTAL   4096
#define L_SEQ     17
#define S_FUTN    16
#define D_MODEL   128
#define D_INF     324
#define D_INNER   256
#define D_STATE   32
#define NHEADS    32
#define HEADDIM   8
#define D_CONV    4
#define CONV_DIM  320
#define D_IN_PROJ 608
#define N_LAYER   8
#define EPSF      1e-5f

#define KPAD_IN   352           // 324 padded to 11 * 32 (embed GEMM K)
#define NPAD_OUT  336           // 324 padded to 21 * 16 (output GEMM N)

#define BT    4                 // batches per workgroup in layer kernel
#define MROWS (BT * L_SEQ)      // 68 valid token rows
#define MPAD  80                // padded to 5 WMMA M-tiles

typedef unsigned short bf16_t;  // storage type (host-safe)
typedef __attribute__((ext_vector_type(16))) __bf16 v16bf;
typedef __attribute__((ext_vector_type(8)))  __bf16 v8bf;
typedef __attribute__((ext_vector_type(8)))  float  v8f;
typedef __attribute__((ext_vector_type(4)))  unsigned int v4u;
typedef __attribute__((ext_vector_type(8)))  int v8i;
typedef __attribute__((ext_vector_type(4)))  int v4i;

__device__ __forceinline__ bf16_t f2bf(float f) {
    unsigned u = __builtin_bit_cast(unsigned, f);
    unsigned r = (u + 0x7FFFu + ((u >> 16) & 1u)) >> 16;
    return (bf16_t)r;
}
__device__ __forceinline__ float bf2f(bf16_t b) {
    unsigned u = ((unsigned)b) << 16;
    return __builtin_bit_cast(float, u);
}
__device__ __forceinline__ float sigmoidf_(float x) { return 1.0f / (1.0f + expf(-x)); }

// A-fragment loader: per ISA §7.12.2, lane holds K chunks [c0,c0+8) and [c0+16,c0+24)
__device__ __forceinline__ v16bf load_afrag(const bf16_t* p) {
    v8bf alo = *(const v8bf*)p;
    v8bf ahi = *(const v8bf*)(p + 16);
    v16bf af;
    #pragma unroll
    for (int i = 0; i < 8; ++i) { af[i] = alo[i]; af[i + 8] = ahi[i]; }
    return af;
}

// Tensor DMA descriptor (D#) builder: fp32 elements, 2-D tile (use dim1=1,tile1=0 for 1-D).
__device__ __forceinline__ void tdm_fill(v4u& g0, v8i& g1,
                                         unsigned long long ga, unsigned lds,
                                         unsigned dim0, unsigned dim1,
                                         unsigned tile0, unsigned tile1,
                                         unsigned stride0) {
    g0[0] = 1u;                                                  // count=1 valid descriptor
    g0[1] = lds;                                                 // lds_addr
    g0[2] = (unsigned)(ga & 0xFFFFFFFFull);                      // global_addr[31:0]
    g0[3] = (unsigned)((ga >> 32) & 0x01FFFFFFull) | (2u << 30); // global_addr[56:32], type=2
    g1[0] = (int)(2u << 16);                                     // data_size=4B, no mask/flags
    g1[1] = (int)((dim0 & 0xFFFFu) << 16);                       // tensor_dim0[15:0]
    g1[2] = (int)(((dim0 >> 16) & 0xFFFFu) | ((dim1 & 0xFFFFu) << 16));
    g1[3] = (int)(((dim1 >> 16) & 0xFFFFu) | ((tile0 & 0xFFFFu) << 16));
    g1[4] = (int)(tile1 & 0xFFFFu);                              // tile_dim1 (tile_dim2=0)
    g1[5] = (int)stride0;                                        // tensor_dim0_stride lo
    g1[6] = 0; g1[7] = 0;
}

// ---------------------------------------------------------------- weight converts
__global__ void k_convert_bf16(const float* __restrict__ src, bf16_t* __restrict__ dst, int n) {
    int i = blockIdx.x * blockDim.x + threadIdx.x;
    if (i < n) dst[i] = f2bf(src[i]);
}
// input_w (64,324) -> (64,352) bf16, K zero-padded
__global__ void k_convert_pad_embedw(const float* __restrict__ src, bf16_t* __restrict__ dst) {
    int i = blockIdx.x * blockDim.x + threadIdx.x;
    if (i >= 64 * KPAD_IN) return;
    int j = i / KPAD_IN, k = i % KPAD_IN;
    dst[i] = (k < D_INF) ? f2bf(src[j * D_INF + k]) : (bf16_t)0;
}
// output_w (324,128) -> (336,128) bf16, N zero-padded rows
__global__ void k_convert_pad_outw(const float* __restrict__ src, bf16_t* __restrict__ dst) {
    int i = blockIdx.x * blockDim.x + threadIdx.x;
    if (i >= NPAD_OUT * D_MODEL) return;
    int j = i >> 7;
    dst[i] = (j < D_INF) ? f2bf(src[i]) : (bf16_t)0;
}

// ---------------------------------------------------------------- embed GEMM (WMMA) + te/se scale
// grid: 69632/128 = 544 blocks, 256 threads. Rows R = b*17+t; x[:,64:128] = in @ Wemb^T + b
__global__ __launch_bounds__(256) void k_embed(
    const float* __restrict__ init_s, const float* __restrict__ fut_s,
    const float* __restrict__ tf, const float* __restrict__ sc,
    const bf16_t* __restrict__ WembP,  // (64,352) bf16
    const float* __restrict__ in_b, const float* __restrict__ emb,
    const float* __restrict__ tw, const float* __restrict__ tb,
    const float* __restrict__ sw, const float* __restrict__ sb,
    float* __restrict__ xg) {
    __shared__ __align__(32) bf16_t sA[128 * KPAD_IN];
    __shared__ float sTW[256], sTB[256], sSW[256], sSB[256];
    const int tid = threadIdx.x, lane = tid & 31, wave = tid >> 5;
    const int R0 = blockIdx.x * 128;

    if (tid < 256) { sTW[tid] = tw[tid]; sTB[tid] = tb[tid]; sSW[tid] = sw[tid]; sSB[tid] = sb[tid]; }
    for (int e = tid; e < 128 * KPAD_IN; e += 256) {
        int r = e / KPAD_IN, k = e % KPAD_IN;
        int R = R0 + r, b = R / L_SEQ, t = R % L_SEQ;
        float v = 0.f;
        if (k < D_INF)
            v = (t == 0) ? init_s[(size_t)b * D_INF + k]
                         : fut_s[((size_t)b * S_FUTN + (t - 1)) * D_INF + k];
        sA[e] = f2bf(v);
    }
    __syncthreads();

    const int hi = lane >> 4, lo = lane & 15;
    for (int tile = wave; tile < 8 * 4; tile += 8) {   // M=128(8 tiles), N=64(4 tiles)
        int mt = tile & 7, nt = tile >> 3;
        int arow = mt * 16 + lo, jcol = nt * 16 + lo;
        v8f acc = {0.f, 0.f, 0.f, 0.f, 0.f, 0.f, 0.f, 0.f};
        for (int kt = 0; kt < KPAD_IN / 32; ++kt) {
            v16bf af = load_afrag(&sA[arow * KPAD_IN + kt * 32 + hi * 8]);
            v16bf bfv = load_afrag(&WembP[jcol * KPAD_IN + kt * 32 + hi * 8]);
            acc = __builtin_amdgcn_wmma_f32_16x16x32_bf16(false, af, false, bfv, (short)0, acc, false, false);
        }
        int d = 64 + nt * 16 + lo;
        #pragma unroll
        for (int v = 0; v < 8; ++v) {
            int R = R0 + mt * 16 + v + hi * 8;
            int b = R / L_SEQ;
            float tfb = tf[b], scb = sc[b];
            float te0 = fmaf(tfb, sTW[d], sTB[d]);
            float se0 = fmaf(scb, sSW[d], sSB[d]);
            float te1 = fmaf(tfb, sTW[128 + d], sTB[128 + d]);
            float se1 = fmaf(scb, sSW[128 + d], sSB[128 + d]);
            float val = acc[v] + in_b[d - 64];
            xg[(size_t)R * D_MODEL + d] = val * te0 * se0 + te1 + se1;
        }
    }
    // emb half: x[:, 0:64]
    for (int e = tid; e < 128 * 64; e += 256) {
        int r = e >> 6, d = e & 63;
        int R = R0 + r, b = R / L_SEQ, t = R % L_SEQ;
        float tfb = tf[b], scb = sc[b];
        float te0 = fmaf(tfb, sTW[d], sTB[d]);
        float se0 = fmaf(scb, sSW[d], sSB[d]);
        float te1 = fmaf(tfb, sTW[128 + d], sTB[128 + d]);
        float se1 = fmaf(scb, sSW[128 + d], sSB[128 + d]);
        xg[(size_t)R * D_MODEL + d] = emb[t * 64 + d] * te0 * se0 + te1 + se1;
    }
}

// ---------------------------------------------------------------- fused mamba2 layer
__global__ __launch_bounds__(1024) void k_layer(
    float* __restrict__ xg,
    const bf16_t* __restrict__ Win,    // (608,128) bf16 row-major
    const bf16_t* __restrict__ Wout,   // (128,256) bf16 row-major
    const float* __restrict__ conv_w,  // (320,4)
    const float* __restrict__ conv_b,  // (320)
    const float* __restrict__ dt_bias, // (32)
    const float* __restrict__ A_log,   // (32)
    const float* __restrict__ D_skip,  // (32)
    const float* __restrict__ nrm_w,   // (256)
    const float* __restrict__ ln_w) {  // (128)
    __shared__ __align__(16) float  sXf[MPAD * D_MODEL];   // TDM-staged x; residual added in place
    __shared__ __align__(32) bf16_t sXn[MPAD * D_MODEL];   // rmsnorm(x) bf16 (A-operand)
    __shared__ __align__(32) bf16_t sZY[MPAD * D_INNER];   // z bf16, later ynorm bf16 (A-operand)
    __shared__ float sXBC[MPAD * CONV_DIM];                // xBC, conv'd in place
    __shared__ float sDT[MPAD * NHEADS];
    __shared__ float sAc[MPAD * NHEADS];
    __shared__ float sY[MPAD * D_INNER];

    const int tid = threadIdx.x;
    const int lane = tid & 31;
    const int wave = tid >> 5;
    const int b0 = blockIdx.x * BT;

    __builtin_prefetch(Win, 0, 2);
    __builtin_prefetch(Wout, 0, 2);

    const unsigned long long ga =
        (unsigned long long)(const void*)(xg + (size_t)b0 * L_SEQ * D_MODEL);
    const unsigned ldsXf = (unsigned)(unsigned long long)(const void*)&sXf[0];
    const unsigned ne = MROWS * D_MODEL;  // 8704 contiguous fp32

    // ---- TDM: async-load the contiguous 68x128 fp32 x tile into LDS (1-D tile)
    if (wave == 0) {
        v4u g0; v8i g1;
        tdm_fill(g0, g1, ga, ldsXf, ne, 1, ne, 0, ne);
        v4i gz4 = {0, 0, 0, 0};
        v8i gz8 = {0, 0, 0, 0, 0, 0, 0, 0};
        __builtin_amdgcn_tensor_load_to_lds(g0, g1, gz4, gz4, gz8, 0);
    }
    // zero pad rows of WMMA A operands (overlaps with TDM)
    for (int i = tid; i < MPAD * D_MODEL; i += 1024) sXn[i] = 0;
    for (int i = tid; i < MPAD * D_INNER; i += 1024) sZY[i] = 0;
    if (wave == 0) __builtin_amdgcn_s_wait_tensorcnt(0);
    __syncthreads();

    // ---- phase 1: rmsnorm(x) -> bf16 A tile. 8 lanes per row, shfl_xor reduction.
    if (tid < MROWS * 8) {
        int m = tid >> 3, li = tid & 7;
        const float* xr = &sXf[m * D_MODEL + li * 16];
        float ss = 0.f;
        #pragma unroll
        for (int dd = 0; dd < 16; ++dd) { float v = xr[dd]; ss = fmaf(v, v, ss); }
        ss += __shfl_xor(ss, 1); ss += __shfl_xor(ss, 2); ss += __shfl_xor(ss, 4);
        float rs = rsqrtf(ss * (1.0f / D_MODEL) + EPSF);
        #pragma unroll
        for (int dd = 0; dd < 16; ++dd) {
            int d = li * 16 + dd;
            sXn[m * D_MODEL + d] = f2bf(sXf[m * D_MODEL + d] * rs * ln_w[d]);
        }
    }
    __syncthreads();

    // ---- phase 2: in_proj GEMM  Y[m,j] = sum_k Xn[m,k] * Win[j,k]   (M=80,N=608,K=128)
    // region boundaries (z:0-255, xBC:256-575, dt:576-607) are 16-aligned -> uniform per tile
    {
        const int MT = MPAD / 16;       // 5
        const int NT = D_IN_PROJ / 16;  // 38
        const int hi = lane >> 4;
        const int lo = lane & 15;
        for (int tile = wave; tile < MT * NT; tile += 32) {
            int mt = tile % MT, nt = tile / MT;
            int arow = mt * 16 + lo;
            int col = nt * 16 + lo;
            v8f acc = {0.f, 0.f, 0.f, 0.f, 0.f, 0.f, 0.f, 0.f};
            for (int kt = 0; kt < D_MODEL / 32; ++kt) {
                v16bf af = load_afrag(&sXn[arow * D_MODEL + kt * 32 + hi * 8]);
                v16bf bfv = load_afrag(&Win[col * D_MODEL + kt * 32 + hi * 8]);
                acc = __builtin_amdgcn_wmma_f32_16x16x32_bf16(false, af, false, bfv, (short)0, acc, false, false);
            }
            if (nt < 16) {                       // z region
                #pragma unroll
                for (int v = 0; v < 8; ++v) {
                    int m = mt * 16 + v + hi * 8;
                    if (m < MROWS) sZY[m * D_INNER + col] = f2bf(acc[v]);
                }
            } else if (nt < 36) {                // xBC region
                int c = col - D_INNER;
                #pragma unroll
                for (int v = 0; v < 8; ++v) {
                    int m = mt * 16 + v + hi * 8;
                    if (m < MROWS) sXBC[m * CONV_DIM + c] = acc[v];
                }
            } else {                             // dt region
                int c = col - (D_INNER + CONV_DIM);
                #pragma unroll
                for (int v = 0; v < 8; ++v) {
                    int m = mt * 16 + v + hi * 8;
                    if (m < MROWS) sDT[m * NHEADS + c] = acc[v];
                }
            }
        }
    }
    __syncthreads();

    // ---- phase 3: dt = softplus(dt + bias), a = exp(-exp(A_log)*dt)
    for (int i = tid; i < MROWS * NHEADS; i += 1024) {
        int h = i % NHEADS;
        float x = sDT[i] + dt_bias[h];
        float dt = (x > 20.f) ? x : log1pf(expf(x));
        sDT[i] = dt;
        sAc[i] = expf(-expf(A_log[h]) * dt);
    }
    // ---- phase 3b: causal depthwise conv + silu, in place per (batch,channel) column
    for (int col = tid; col < BT * CONV_DIM; col += 1024) {
        int bl = col / CONV_DIM, ch = col % CONV_DIM;
        float w0 = conv_w[ch * 4 + 0], w1 = conv_w[ch * 4 + 1];
        float w2 = conv_w[ch * 4 + 2], w3 = conv_w[ch * 4 + 3];
        float cb = conv_b[ch];
        float h0 = 0.f, h1 = 0.f, h2 = 0.f, h3 = 0.f;
        for (int t = 0; t < L_SEQ; ++t) {
            int m = bl * L_SEQ + t;
            h0 = h1; h1 = h2; h2 = h3; h3 = sXBC[m * CONV_DIM + ch];
            float acc = cb + w0 * h0 + w1 * h1 + w2 * h2 + w3 * h3;
            sXBC[m * CONV_DIM + ch] = acc * sigmoidf_(acc);
        }
    }
    __syncthreads();

    // ---- phase 4: SSM scan, thread = (batch, head, p), state (n=32) in VGPRs
    {
        int bl = tid >> 8;
        int h = (tid >> 3) & 31;
        int p = tid & 7;
        float f[D_STATE];
        #pragma unroll
        for (int n = 0; n < D_STATE; ++n) f[n] = 0.f;
        float Dh = D_skip[h];
        for (int t = 0; t < L_SEQ; ++t) {
            int m = bl * L_SEQ + t;
            float xs = sXBC[m * CONV_DIM + h * HEADDIM + p];
            float dt = sDT[m * NHEADS + h];
            float a = sAc[m * NHEADS + h];
            float xd = xs * dt;
            float y = 0.f;
            #pragma unroll
            for (int n = 0; n < D_STATE; ++n) {
                float Bn = sXBC[m * CONV_DIM + D_INNER + n];
                float Cn = sXBC[m * CONV_DIM + D_INNER + D_STATE + n];
                f[n] = a * f[n] + xd * Bn;
                y = fmaf(f[n], Cn, y);
            }
            sY[m * D_INNER + h * HEADDIM + p] = y + Dh * xs;
        }
    }
    __syncthreads();

    // ---- phase 5: gated rmsnorm: ynorm = rmsnorm(y*silu(z))*nrm_w, bf16 into sZY.
    if (tid < MROWS * 8) {
        int m = tid >> 3, li = tid & 7;
        float ss = 0.f;
        #pragma unroll
        for (int cc = 0; cc < 32; ++cc) {
            int c = li * 32 + cc;
            float z = bf2f(sZY[m * D_INNER + c]);
            float val = sY[m * D_INNER + c] * (z * sigmoidf_(z));
            ss = fmaf(val, val, ss);
        }
        ss += __shfl_xor(ss, 1); ss += __shfl_xor(ss, 2); ss += __shfl_xor(ss, 4);
        float rs = rsqrtf(ss * (1.0f / D_INNER) + EPSF);
        #pragma unroll
        for (int cc = 0; cc < 32; ++cc) {
            int c = li * 32 + cc;
            float z = bf2f(sZY[m * D_INNER + c]);
            float val = sY[m * D_INNER + c] * (z * sigmoidf_(z));
            sZY[m * D_INNER + c] = f2bf(val * rs * nrm_w[c]);
        }
    }
    __syncthreads();

    // ---- phase 6: out_proj GEMM (M=80,N=128,K=256); residual added in place into sXf.
    // Each (m,d) element belongs to exactly one tile -> no races.
    {
        const int MT = MPAD / 16;     // 5
        const int NT = D_MODEL / 16;  // 8
        const int hi = lane >> 4;
        const int lo = lane & 15;
        for (int tile = wave; tile < MT * NT; tile += 32) {
            int mt = tile % MT, nt = tile / MT;
            int arow = mt * 16 + lo;
            int dcol = nt * 16 + lo;
            v8f acc = {0.f, 0.f, 0.f, 0.f, 0.f, 0.f, 0.f, 0.f};
            for (int kt = 0; kt < D_INNER / 32; ++kt) {
                v16bf af = load_afrag(&sZY[arow * D_INNER + kt * 32 + hi * 8]);
                v16bf bfv = load_afrag(&Wout[dcol * D_INNER + kt * 32 + hi * 8]);
                acc = __builtin_amdgcn_wmma_f32_16x16x32_bf16(false, af, false, bfv, (short)0, acc, false, false);
            }
            #pragma unroll
            for (int v = 0; v < 8; ++v) {
                int m = mt * 16 + v + hi * 8;
                if (m < MROWS) {
                    int li = m * D_MODEL + dcol;
                    sXf[li] = sXf[li] + acc[v];
                }
            }
        }
    }
    __syncthreads();

    // ---- TDM: async-store the updated contiguous 68x128 fp32 x tile back to global
    if (wave == 0) {
        v4u g0; v8i g1;
        tdm_fill(g0, g1, ga, ldsXf, ne, 1, ne, 0, ne);
        v4i gz4 = {0, 0, 0, 0};
        v8i gz8 = {0, 0, 0, 0, 0, 0, 0, 0};
        __builtin_amdgcn_tensor_store_from_lds(g0, g1, gz4, gz4, gz8, 0);
        __builtin_amdgcn_s_wait_tensorcnt(0);
    }
}

// ---------------------------------------------------------------- final scale + output proj (WMMA)
// grid: 65536/128 = 512 blocks, 256 threads. Rows R = b*16+tt, token t = tt+1.
// Each block covers exactly 8 batches; tokens 1..16 of a batch are a contiguous
// 2048-float run with batch stride 2176 floats -> 2-D TDM tile.
__global__ __launch_bounds__(256) void k_output(
    const float* __restrict__ xg,
    const float* __restrict__ tf, const float* __restrict__ sc,
    const float* __restrict__ tw, const float* __restrict__ tb,
    const float* __restrict__ sw, const float* __restrict__ sb,
    const bf16_t* __restrict__ WoutP,  // (336,128) bf16 padded
    const float* __restrict__ out_b, float* __restrict__ out) {
    __shared__ __align__(16) float  sXf[128 * D_MODEL];   // TDM-staged x rows
    __shared__ __align__(32) bf16_t sA[128 * D_MODEL];
    __shared__ float sTW[256], sTB[256], sSW[256], sSB[256];
    const int tid = threadIdx.x, lane = tid & 31, wave = tid >> 5;
    const int R0 = blockIdx.x * 128;
    const int bbase = R0 >> 4;   // 8 batches per block

    if (wave == 0) {
        unsigned long long ga = (unsigned long long)(const void*)
            (xg + ((size_t)bbase * L_SEQ + 1) * D_MODEL);
        unsigned lds = (unsigned)(unsigned long long)(const void*)&sXf[0];
        v4u g0; v8i g1;
        tdm_fill(g0, g1, ga, lds,
                 /*dim0=*/S_FUTN * D_MODEL, /*dim1=*/8,
                 /*tile0=*/S_FUTN * D_MODEL, /*tile1=*/8,
                 /*stride0=*/L_SEQ * D_MODEL);
        v4i gz4 = {0, 0, 0, 0};
        v8i gz8 = {0, 0, 0, 0, 0, 0, 0, 0};
        __builtin_amdgcn_tensor_load_to_lds(g0, g1, gz4, gz4, gz8, 0);
    }
    if (tid < 256) {
        sTW[tid] = tw[256 + tid]; sTB[tid] = tb[256 + tid];
        sSW[tid] = sw[256 + tid]; sSB[tid] = sb[256 + tid];
    }
    if (wave == 0) __builtin_amdgcn_s_wait_tensorcnt(0);
    __syncthreads();

    for (int e = tid; e < 128 * D_MODEL; e += 256) {
        int r = e >> 7, d = e & 127;
        int R = R0 + r, b = R >> 4;
        float xv = sXf[e];   // sXf[(bl*16+tt)*128+d] == sXf[r*128+d]
        float tfb = tf[b], scb = sc[b];
        float te2 = fmaf(tfb, sTW[d], sTB[d]);
        float se2 = fmaf(scb, sSW[d], sSB[d]);
        float te3 = fmaf(tfb, sTW[128 + d], sTB[128 + d]);
        float se3 = fmaf(scb, sSW[128 + d], sSB[128 + d]);
        sA[e] = f2bf(xv * te2 * se2 + te3 + se3);
    }
    __syncthreads();

    const int hi = lane >> 4, lo = lane & 15;
    for (int tile = wave; tile < 8 * (NPAD_OUT / 16); tile += 8) {  // 8 x 21 tiles
        int mt = tile & 7, nt = tile >> 3;
        int arow = mt * 16 + lo;
        int j = nt * 16 + lo;
        v8f acc = {0.f, 0.f, 0.f, 0.f, 0.f, 0.f, 0.f, 0.f};
        for (int kt = 0; kt < D_MODEL / 32; ++kt) {
            v16bf af = load_afrag(&sA[arow * D_MODEL + kt * 32 + hi * 8]);
            v16bf bfv = load_afrag(&WoutP[j * D_MODEL + kt * 32 + hi * 8]);
            acc = __builtin_amdgcn_wmma_f32_16x16x32_bf16(false, af, false, bfv, (short)0, acc, false, false);
        }
        if (j < D_INF) {
            float bj = out_b[j];
            #pragma unroll
            for (int v = 0; v < 8; ++v) {
                int R = R0 + mt * 16 + v + hi * 8;
                out[(size_t)R * D_INF + j] = acc[v] + bj;
            }
        }
    }
}

// ---------------------------------------------------------------- host entry
extern "C" void kernel_launch(void* const* d_in, const int* in_sizes, int n_in,
                              void* d_out, int out_size, void* d_ws, size_t ws_size,
                              hipStream_t stream) {
    const float* init_s  = (const float*)d_in[0];
    const float* fut_s   = (const float*)d_in[1];
    const float* tflag   = (const float*)d_in[2];
    const float* sval    = (const float*)d_in[3];
    const float* in_w    = (const float*)d_in[4];
    const float* in_b    = (const float*)d_in[5];
    const float* emb     = (const float*)d_in[6];
    const float* tw      = (const float*)d_in[7];
    const float* tb      = (const float*)d_in[8];
    const float* sw      = (const float*)d_in[9];
    const float* sb      = (const float*)d_in[10];
    const float* inproj  = (const float*)d_in[11];
    const float* convw   = (const float*)d_in[12];
    const float* convb   = (const float*)d_in[13];
    const float* dtb     = (const float*)d_in[14];
    const float* alog    = (const float*)d_in[15];
    const float* dskip   = (const float*)d_in[16];
    const float* mnw     = (const float*)d_in[17];
    const float* outproj = (const float*)d_in[18];
    const float* lnw     = (const float*)d_in[19];
    const float* outw    = (const float*)d_in[20];
    const float* outb    = (const float*)d_in[21];
    float* out = (float*)d_out;

    char* ws = (char*)d_ws;
    float* xg = (float*)ws;  // (B,17,128) fp32 activations
    size_t off = ((size_t)B_TOTAL * L_SEQ * D_MODEL * sizeof(float) + 255) & ~(size_t)255;
    bf16_t* WinB = (bf16_t*)(ws + off);
    size_t winN = (size_t)N_LAYER * D_IN_PROJ * D_MODEL;
    off = (off + winN * sizeof(bf16_t) + 255) & ~(size_t)255;
    bf16_t* WoutB = (bf16_t*)(ws + off);
    size_t woutN = (size_t)N_LAYER * D_MODEL * D_INNER;
    off = (off + woutN * sizeof(bf16_t) + 255) & ~(size_t)255;
    bf16_t* WembP = (bf16_t*)(ws + off);
    size_t wembN = (size_t)64 * KPAD_IN;
    off = (off + wembN * sizeof(bf16_t) + 255) & ~(size_t)255;
    bf16_t* WoutP = (bf16_t*)(ws + off);
    size_t woutpN = (size_t)NPAD_OUT * D_MODEL;

    k_convert_bf16<<<(unsigned)((winN + 255) / 256), 256, 0, stream>>>(inproj, WinB, (int)winN);
    k_convert_bf16<<<(unsigned)((woutN + 255) / 256), 256, 0, stream>>>(outproj, WoutB, (int)woutN);
    k_convert_pad_embedw<<<(unsigned)((wembN + 255) / 256), 256, 0, stream>>>(in_w, WembP);
    k_convert_pad_outw<<<(unsigned)((woutpN + 255) / 256), 256, 0, stream>>>(outw, WoutP);

    k_embed<<<(B_TOTAL * L_SEQ) / 128, 256, 0, stream>>>(init_s, fut_s, tflag, sval, WembP,
                                                         in_b, emb, tw, tb, sw, sb, xg);

    for (int i = 0; i < N_LAYER; ++i) {
        k_layer<<<B_TOTAL / BT, 1024, 0, stream>>>(
            xg,
            WinB + (size_t)i * D_IN_PROJ * D_MODEL,
            WoutB + (size_t)i * D_MODEL * D_INNER,
            convw + i * CONV_DIM * D_CONV, convb + i * CONV_DIM,
            dtb + i * NHEADS, alog + i * NHEADS, dskip + i * NHEADS,
            mnw + i * D_INNER, lnw + i * D_MODEL);
    }

    k_output<<<(B_TOTAL * S_FUTN) / 128, 256, 0, stream>>>(xg, tflag, sval, tw, tb, sw, sb,
                                                           WoutP, outb, out);
}
